// JointRecStatic_7301444403200
// MI455X (gfx1250) — matrix operand order repaired and checked by reading, not compiled, not run on hardware
//
#include <hip/hip_runtime.h>
#include <hip/hip_bf16.h>

#define N_USER 100000
#define M_ITEM 50000
#define D 64
#define E_EDGES 1000000
#define BB 2048
#define LH 50
#define DEPTH 3
#define TAU 0.2f

typedef __attribute__((ext_vector_type(16))) _Float16 v16h;
typedef __attribute__((ext_vector_type(8)))  float    v8f;

__device__ __forceinline__ float softplusf(float x) {
    return x > 20.f ? x : log1pf(expf(x));
}

__device__ __forceinline__ float wave_sum(float v) {
    // wave32 butterfly reduction
    for (int off = 16; off > 0; off >>= 1)
        v += __shfl_xor(v, off, 32);
    return v;
}

// ---------------------------------------------------------------- utilities
__global__ void fill_zero_kernel(float* p, long n) {
    long i = (long)blockIdx.x * blockDim.x + threadIdx.x;
    long stride = (long)gridDim.x * blockDim.x;
    for (; i < n; i += stride) p[i] = 0.f;
}

__global__ void copy2_kernel(float* d1, float* d2, const float* s, long n) {
    long i = (long)blockIdx.x * blockDim.x + threadIdx.x;
    long stride = (long)gridDim.x * blockDim.x;
    for (; i < n; i += stride) { float v = s[i]; d1[i] = v; d2[i] = v; }
}

__global__ void accum_kernel(float* acc, const float* x, long n) {
    long i = (long)blockIdx.x * blockDim.x + threadIdx.x;
    long stride = (long)gridDim.x * blockDim.x;
    for (; i < n; i += stride) acc[i] += x[i];
}

// ---------------------------------------------------------------- degrees & norm
__global__ void degree_kernel(const int* __restrict__ eu, const int* __restrict__ ei,
                              float* deg_u, float* deg_i) {
    int e = blockIdx.x * blockDim.x + threadIdx.x;
    if (e >= E_EDGES) return;
    atomicAdd(&deg_u[eu[e]], 1.0f);
    atomicAdd(&deg_i[ei[e]], 1.0f);
}

__global__ void norm_kernel(const int* __restrict__ eu, const int* __restrict__ ei,
                            const float* __restrict__ deg_u, const float* __restrict__ deg_i,
                            float* __restrict__ nrm) {
    int e = blockIdx.x * blockDim.x + threadIdx.x;
    if (e >= E_EDGES) return;
    float du = fmaxf(deg_u[eu[e]], 1.0f);
    float di = fmaxf(deg_i[ei[e]], 1.0f);
    nrm[e] = rsqrtf(du * di);
}

// ---------------------------------------------------------------- edge scatter (one wave per edge, 2 dims/lane)
__global__ void scatter_kernel(const int* __restrict__ eu, const int* __restrict__ ei,
                               const float* __restrict__ nrm,
                               const float* __restrict__ ucur, const float* __restrict__ icur,
                               float* unext, float* inext) {
    int wave = (blockIdx.x * blockDim.x + threadIdx.x) >> 5;
    int lane = threadIdx.x & 31;
    if (wave >= E_EDGES) return;
    int u  = eu[wave];
    int it = ei[wave];
    float nz = nrm[wave];
    const float* ic = icur + (size_t)it * D;
    const float* uc = ucur + (size_t)u  * D;
    float* un = unext + (size_t)u  * D;
    float* in = inext + (size_t)it * D;
#pragma unroll
    for (int k = 0; k < 2; k++) {
        int d = lane + k * 32;
        atomicAdd(&un[d], ic[d] * nz);
        atomicAdd(&in[d], uc[d] * nz);
    }
}

// ---------------------------------------------------------------- gather final embeddings for batch
__global__ void gather_kernel(const int* __restrict__ users, const int* __restrict__ items,
                              const float* __restrict__ uacc, const float* __restrict__ iacc,
                              float* uf, float* vf, _Float16* ufh, _Float16* vfh) {
    int t = blockIdx.x * blockDim.x + threadIdx.x;
    if (t >= BB * D) return;
    int b = t >> 6, d = t & 63;
    const float inv = 1.0f / (DEPTH + 1);
    float u = uacc[(size_t)users[b] * D + d] * inv;
    float v = iacc[(size_t)items[b] * D + d] * inv;
    uf[t] = u; vf[t] = v;
    ufh[t] = (_Float16)u; vfh[t] = (_Float16)v;
}

// ---------------------------------------------------------------- Hawkes intensity (one wave per event)
__global__ void hawkes_kernel(const float* __restrict__ uf, const float* __restrict__ vf,
                              const float* __restrict__ iacc,
                              const int* __restrict__ hist_items,
                              const float* __restrict__ event_time,
                              const float* __restrict__ hist_time,
                              const float* __restrict__ decay_raw,
                              float* acc) {
    int wave = (blockIdx.x * blockDim.x + threadIdx.x) >> 5;
    int lane = threadIdx.x & 31;
    if (wave >= BB) return;
    float decay = softplusf(decay_raw[0]);
    const float* u = uf + (size_t)wave * D;
    const float* v = vf + (size_t)wave * D;
    float v0 = v[lane], v1 = v[lane + 32];
    float base = wave_sum(u[lane] * v0 + u[lane + 32] * v1);
    float et = event_time[wave];
    const float inv = 1.0f / (DEPTH + 1);
    float s = 0.f;
    for (int l = 0; l < LH; l++) {
        int it = hist_items[wave * LH + l];          // uniform per wave
        if (it < M_ITEM) {
            const float* h = iacc + (size_t)it * D;
            float e = wave_sum(h[lane] * v0 + h[lane + 32] * v1) * inv;
            float dt = fmaxf(et - hist_time[wave * LH + l], 0.f);
            s += expf(-decay * dt) * e;
        }
    }
    if (lane == 0) {
        float intensity = softplusf(base + s);
        atomicAdd(&acc[0], -logf(intensity + 1e-8f));
    }
}

// ---------------------------------------------------------------- logits = (u_f @ v_f^T) / tau  via WMMA
// One wave computes a 16x16 f32 tile; K=64 in two 16x16x32 f16 WMMA steps.
__global__ void logits_wmma_kernel(const _Float16* __restrict__ A,
                                   const _Float16* __restrict__ Bm,
                                   float* __restrict__ S) {
    int gwave = (blockIdx.x * blockDim.x + threadIdx.x) >> 5;
    int lane  = threadIdx.x & 31;
    const int TILES = BB / 16;                 // 128 tiles per dim
    int ti = gwave / TILES, tj = gwave % TILES;
    if (ti >= TILES) return;

    int m  = lane & 15;                        // row within A tile / col within B tile
    int hi = lane >> 4;                        // lane-half selects K grouping
    const _Float16* pa = A  + (size_t)(ti * 16 + m) * D;
    const _Float16* pb = Bm + (size_t)(tj * 16 + m) * D;

    v8f c = {};
#pragma unroll
    for (int ks = 0; ks < 2; ks++) {
        int kbase = ks * 32;
        v16h av, bv;
#pragma unroll
        for (int v = 0; v < 8; v++) {
            // A 16x32 f16 layout: lanes 0-15 K={0-7,16-23}, lanes 16-31 K={8-15,24-31}
            int ka = kbase + (v >> 2) * 16 + hi * 8 + (v & 3) * 2;
            av[2 * v]     = pa[ka];
            av[2 * v + 1] = pa[ka + 1];
            // B 32x16 f16 layout: lanes 0-15 K={0-7,16-23}, lanes 16-31 K={8-15,24-31} (col = lane&15)
            int kb = kbase + (v >> 2) * 16 + hi * 8 + (v & 3) * 2;
            bv[2 * v]     = pb[kb];
            bv[2 * v + 1] = pb[kb + 1];
        }
        c = __builtin_amdgcn_wmma_f32_16x16x32_f16(false, av, false, bv,
                                                   (short)0, c, false, false);
    }

    // C/D layout: VGPR r -> M=r (lanes 0-15) / M=r+8 (lanes 16-31); N = lane&15
    const float sc = 1.0f / TAU;
    int row0 = ti * 16 + hi * 8;
    int col  = tj * 16 + m;
#pragma unroll
    for (int r = 0; r < 8; r++)
        S[(size_t)(row0 + r) * BB + col] = c[r] * sc;
}

// ---------------------------------------------------------------- per-row logsumexp, diag NCE loss
__global__ void nce_row_kernel(const float* __restrict__ S, float* acc) {
    int b = blockIdx.x;
    __shared__ float red[256];
    const float* row = S + (size_t)b * BB;
    float m = -3.4e38f;
    for (int j = threadIdx.x; j < BB; j += 256) m = fmaxf(m, row[j]);
    red[threadIdx.x] = m; __syncthreads();
    for (int s = 128; s > 0; s >>= 1) {
        if (threadIdx.x < s) red[threadIdx.x] = fmaxf(red[threadIdx.x], red[threadIdx.x + s]);
        __syncthreads();
    }
    m = red[0]; __syncthreads();
    float sum = 0.f;
    for (int j = threadIdx.x; j < BB; j += 256) sum += expf(row[j] - m);
    red[threadIdx.x] = sum; __syncthreads();
    for (int s = 128; s > 0; s >>= 1) {
        if (threadIdx.x < s) red[threadIdx.x] += red[threadIdx.x + s];
        __syncthreads();
    }
    if (threadIdx.x == 0) {
        float lse = m + logf(red[0]);
        atomicAdd(&acc[1], lse - row[b]);    // -log_softmax diagonal
    }
}

__global__ void finalize_kernel(const float* __restrict__ acc, float* out) {
    if (blockIdx.x == 0 && threadIdx.x == 0)
        out[0] = (acc[0] + acc[1]) * (1.0f / BB);
}

// ---------------------------------------------------------------- host driver
extern "C" void kernel_launch(void* const* d_in, const int* in_sizes, int n_in,
                              void* d_out, int out_size, void* d_ws, size_t ws_size,
                              hipStream_t stream) {
    (void)in_sizes; (void)n_in; (void)out_size;
    const float* user_emb   = (const float*)d_in[0];
    const float* item_emb   = (const float*)d_in[1];
    const float* decay_raw  = (const float*)d_in[2];
    const float* event_time = (const float*)d_in[3];
    const float* hist_time  = (const float*)d_in[4];
    const int*   edge_user  = (const int*)d_in[5];
    const int*   edge_item  = (const int*)d_in[6];
    const int*   users      = (const int*)d_in[7];
    const int*   pos_items  = (const int*)d_in[8];
    const int*   hist_items = (const int*)d_in[9];
    float* out = (float*)d_out;

    char* p = (char*)d_ws;
    auto carve = [&](size_t bytes) -> char* {
        char* r = p;
        p += (bytes + 255) & ~(size_t)255;
        return r;
    };
    float*    deg_u = (float*)carve((size_t)N_USER * 4);
    float*    deg_i = (float*)carve((size_t)M_ITEM * 4);
    float*    nrm   = (float*)carve((size_t)E_EDGES * 4);
    float*    u_a   = (float*)carve((size_t)N_USER * D * 4);
    float*    u_b   = (float*)carve((size_t)N_USER * D * 4);
    float*    u_acc = (float*)carve((size_t)N_USER * D * 4);
    float*    i_a   = (float*)carve((size_t)M_ITEM * D * 4);
    float*    i_b   = (float*)carve((size_t)M_ITEM * D * 4);
    float*    i_acc = (float*)carve((size_t)M_ITEM * D * 4);
    float*    uf    = (float*)carve((size_t)BB * D * 4);
    float*    vf    = (float*)carve((size_t)BB * D * 4);
    _Float16* ufh   = (_Float16*)carve((size_t)BB * D * 2);
    _Float16* vfh   = (_Float16*)carve((size_t)BB * D * 2);
    float*    logits = (float*)carve((size_t)BB * BB * 4);
    float*    acc   = (float*)carve(256);
    if ((size_t)(p - (char*)d_ws) > ws_size) return;   // workspace too small

    const int T = 256;
    // zero everything we accumulate into (graph-replay safe)
    fill_zero_kernel<<<512,  T, 0, stream>>>(deg_u, N_USER);
    fill_zero_kernel<<<512,  T, 0, stream>>>(deg_i, M_ITEM);
    fill_zero_kernel<<<1,    T, 0, stream>>>(acc, 64);

    degree_kernel<<<(E_EDGES + T - 1) / T, T, 0, stream>>>(edge_user, edge_item, deg_u, deg_i);
    norm_kernel  <<<(E_EDGES + T - 1) / T, T, 0, stream>>>(edge_user, edge_item, deg_u, deg_i, nrm);

    copy2_kernel<<<2048, T, 0, stream>>>(u_a, u_acc, user_emb, (long)N_USER * D);
    copy2_kernel<<<2048, T, 0, stream>>>(i_a, i_acc, item_emb, (long)M_ITEM * D); // first M_ITEM rows

    float* ucur = u_a; float* unxt = u_b;
    float* icur = i_a; float* inxt = i_b;
    for (int layer = 0; layer < DEPTH; layer++) {
        fill_zero_kernel<<<4096, T, 0, stream>>>(unxt, (long)N_USER * D);
        fill_zero_kernel<<<4096, T, 0, stream>>>(inxt, (long)M_ITEM * D);
        scatter_kernel<<<(E_EDGES * 32 + T - 1) / T, T, 0, stream>>>(
            edge_user, edge_item, nrm, ucur, icur, unxt, inxt);
        accum_kernel<<<4096, T, 0, stream>>>(u_acc, unxt, (long)N_USER * D);
        accum_kernel<<<4096, T, 0, stream>>>(i_acc, inxt, (long)M_ITEM * D);
        float* t;
        t = ucur; ucur = unxt; unxt = t;
        t = icur; icur = inxt; inxt = t;
    }

    gather_kernel<<<(BB * D + T - 1) / T, T, 0, stream>>>(
        users, pos_items, u_acc, i_acc, uf, vf, ufh, vfh);

    hawkes_kernel<<<(BB * 32 + T - 1) / T, T, 0, stream>>>(
        uf, vf, i_acc, hist_items, event_time, hist_time, decay_raw, acc);

    // 128x128 = 16384 tiles, 8 waves (tiles) per 256-thread block
    logits_wmma_kernel<<<2048, T, 0, stream>>>(ufh, vfh, logits);

    nce_row_kernel<<<BB, T, 0, stream>>>(logits, acc);
    finalize_kernel<<<1, 64, 0, stream>>>(acc, out);
}
